// RNN_18287970747088
// MI455X (gfx1250) — matrix operand order repaired and checked by reading, not compiled, not run on hardware
//
#include <hip/hip_runtime.h>

// ---------------------------------------------------------------------------
// CTRNN forward for MI455X (gfx1250, wave32).
//   - 16 persistent workgroups, each owns a 16-row batch tile for all T steps
//   - fp32 state x1/x2/x1b lives in WMMA f32 accumulator registers (v8f/tile)
//   - recurrent GEMMs: v_wmma_f32_16x16x32_bf16, weights streamed from L2 in
//     WMMA-B layout (16 contiguous bf16 per lane per fragment).
//     A laundered 32-bit scalar offset (not the pointer!) defeats LICM so the
//     80 fragments are re-read from L2 each step via global_load_b128
//     (saddr form, addrspace(1) preserved) instead of being hoisted+spilled
//     (round 1) or degraded to flat_load with 64-bit VALU adds (round 2).
//   - input projection (K=4): v_wmma_f32_16x16x4_f32, exact fp32
//   - tanh activations double-buffered in LDS as bf16 (padded rows), reloaded
//     as WMMA-A fragments via ds_load_b128; one barrier per step
// ---------------------------------------------------------------------------

typedef __attribute__((ext_vector_type(8)))  float          v8f;
typedef __attribute__((ext_vector_type(2)))  float          v2f;
typedef __attribute__((ext_vector_type(16))) __bf16         v16bf;
typedef __attribute__((ext_vector_type(8)))  unsigned short v8us;
typedef __attribute__((ext_vector_type(16))) unsigned short v16us;

#define T_STEPS 1000
#define BATCH   256
#define NH      256
#define ALPHA_C 0.2f
#define LDSP    264   // padded LDS row stride (elements) -> 528B rows, no bank conflicts

__device__ __forceinline__ unsigned short f2bf(float f) {
  unsigned u = __float_as_uint(f);
  return (unsigned short)((u + 0x7FFFu + ((u >> 16) & 1u)) >> 16);  // RNE
}

__device__ __forceinline__ float fast_tanh(float x) {
  float e = __expf(2.0f * x);
  return 1.0f - 2.0f / (e + 1.0f);
}

// WMMA wrappers ------------------------------------------------------------
__device__ __forceinline__ v8f wmma_bf16(v16bf a, v16bf b, v8f c) {
  return __builtin_amdgcn_wmma_f32_16x16x32_bf16(false, a, false, b,
                                                 (short)0, c, false, false);
}
__device__ __forceinline__ v8f wmma_f32k4(v2f a, v2f b, v8f c) {
  return __builtin_amdgcn_wmma_f32_16x16x4_f32(false, a, false, b,
                                               (short)0, c, false, false);
}

// A fragment (16x32 bf16) from LDS: elements 0..15 = K{hi*16..hi*16+15}
__device__ __forceinline__ v16bf ldsA(const unsigned short* base, int lane, int kt) {
  int row = lane & 15;
  int kb  = (lane >> 4) * 16;
  const unsigned short* p = base + row * LDSP + kt * 32 + kb;
  v8us lo = *(const v8us*)(p);        // 16B chunk: K = kt*32 + kb + [0..7]
  v8us hi = *(const v8us*)(p + 8);    // 16B chunk: K = kt*32 + kb + 8 + [0..7]
  v16us c = __builtin_shufflevector(lo, hi, 0,1,2,3,4,5,6,7,8,9,10,11,12,13,14,15);
  return __builtin_bit_cast(v16bf, c);
}

// B fragment (32x16 bf16) from global W (row-major NxK): lane -> W[n][k..k+15]
// `off` carries matrix base + laundered per-step offset (keeps addrspace(1)).
__device__ __forceinline__ v16bf gB(const unsigned short* __restrict__ Wbase,
                                    unsigned off, int lane, int kt, int ntbase) {
  int n = ntbase + (lane & 15);
  int k = kt * 32 + (lane >> 4) * 16;
  v16us c = *(const v16us*)(Wbase + off + n * NH + k);  // one contiguous 32B load
  return __builtin_bit_cast(v16bf, c);
}

// ---------------------------------------------------------------------------
__global__ void __launch_bounds__(256, 1) ctrnn_kernel(
    const float* __restrict__ X,     const float* __restrict__ h01,
    const float* __restrict__ h02,   const float* __restrict__ h01b,
    const float* __restrict__ Wih0,  const float* __restrict__ Wih0b,
    const unsigned short* __restrict__ Wbf,   // bf16: Whh0,W0bto0,Whh0b,Whh1,Wih1
    float* __restrict__ hid1, float* __restrict__ hid2, float* __restrict__ hid1b)
{
  // double-buffered tanh activations: [buf][pop: r1,r2,r1b][row][padded col]
  __shared__ __align__(16) unsigned short rbuf[2][3][16 * LDSP];

  const int tid  = threadIdx.x;
  const int lane = tid & 31;
  const int w    = tid >> 5;          // wave 0..7, owns column tiles {2w, 2w+1}
  const int m0   = blockIdx.x * 16;   // batch tile base row
  const int hi   = lane >> 4;
  const int ln   = lane & 15;

  // ---- init: load x state into accumulator layout, seed r = tanh(h0) ------
  v8f x1[2], x2[2], x1b[2];
  for (int nti = 0; nti < 2; ++nti) {
    int ntb = (2 * w + nti) * 16;
    int col = ntb + ln;
    for (int j = 0; j < 8; ++j) {
      int lr  = j + 8 * hi;                    // local row (0..15)
      int gr  = m0 + lr;                       // global batch row
      float a = h01 [gr * NH + col];
      float b = h02 [gr * NH + col];
      float c = h01b[gr * NH + col];
      x1 [nti][j] = a;  x2[nti][j] = b;  x1b[nti][j] = c;
      rbuf[0][0][lr * LDSP + col] = f2bf(fast_tanh(a));
      rbuf[0][1][lr * LDSP + col] = f2bf(fast_tanh(b));
      rbuf[0][2][lr * LDSP + col] = f2bf(fast_tanh(c));
    }
  }

  // hoist constant fp32 B-fragments for the K=4 input projections (small)
  v2f Bih0[2], Bih0b[2];
  for (int nti = 0; nti < 2; ++nti) {
    int n  = (2 * w + nti) * 16 + ln;
    int kk = hi * 2;
    Bih0 [nti] = *(const v2f*)(Wih0  + n * 4 + kk);
    Bih0b[nti] = *(const v2f*)(Wih0b + n * 4 + kk);
  }

  __syncthreads();

  const float al = ALPHA_C, oma = 1.0f - ALPHA_C;

  for (int t = 0; t < T_STEPS; ++t) {
    const int cur = t & 1, nxt = cur ^ 1;

    // Laundered scalar offset (always 0 at runtime): makes every weight-load
    // address loop-variant so LICM cannot hoist the 640-VGPR fragment set,
    // while keeping Wbf's global address space (=> global_load_b128 saddr).
    unsigned toff = 0;
    asm volatile("" : "+s"(toff));
    const unsigned o_hh0  = toff + 0 * 65536;
    const unsigned o_0b0  = toff + 1 * 65536;
    const unsigned o_hh0b = toff + 2 * 65536;
    const unsigned o_hh1  = toff + 3 * 65536;
    const unsigned o_ih1  = toff + 4 * 65536;

    // xin A fragment (16x4 fp32): lane -> X[t, m0+ln, hi*2 .. +1]
    v2f xa = *(const v2f*)(X + ((size_t)t * BATCH + m0 + ln) * 4 + hi * 2);

    // two independent accumulation chains per tile (halves WMMA RAW depth)
    v8f c1a[2], c1b[2], c1bb[2], c2a[2], c2b[2];
    for (int nti = 0; nti < 2; ++nti) {
      v8f z = {};
      c1a [nti] = wmma_f32k4(xa, Bih0 [nti], z);   // xin @ W_ih0.T
      c1bb[nti] = wmma_f32k4(xa, Bih0b[nti], z);   // xin @ W_ih0b.T
      c1b [nti] = z;  c2a[nti] = z;  c2b[nti] = z;
    }

    #pragma unroll
    for (int kt = 0; kt < 8; ++kt) {
      v16bf A1  = ldsA(&rbuf[cur][0][0], lane, kt);   // r1
      v16bf A2  = ldsA(&rbuf[cur][1][0], lane, kt);   // r2
      v16bf A1b = ldsA(&rbuf[cur][2][0], lane, kt);   // r1b
      for (int nti = 0; nti < 2; ++nti) {
        int ntb = (2 * w + nti) * 16;
        c1a [nti] = wmma_bf16(A1,  gB(Wbf, o_hh0,  lane, kt, ntb), c1a [nti]);
        c1b [nti] = wmma_bf16(A1b, gB(Wbf, o_0b0,  lane, kt, ntb), c1b [nti]);
        c1bb[nti] = wmma_bf16(A1b, gB(Wbf, o_hh0b, lane, kt, ntb), c1bb[nti]);
        c2a [nti] = wmma_bf16(A2,  gB(Wbf, o_hh1,  lane, kt, ntb), c2a [nti]);
        c2b [nti] = wmma_bf16(A1,  gB(Wbf, o_ih1,  lane, kt, ntb), c2b [nti]);
      }
    }

    // alpha-blend update, tanh, publish activations (LDS bf16 + global f32)
    size_t obase = ((size_t)t * BATCH + m0) * NH;
    for (int nti = 0; nti < 2; ++nti) {
      int col = (2 * w + nti) * 16 + ln;
      for (int j = 0; j < 8; ++j) {
        int    lr   = j + 8 * hi;
        size_t gidx = obase + (size_t)lr * NH + col;

        float xn1 = oma * x1[nti][j] + al * (c1a[nti][j] + c1b[nti][j]);
        x1[nti][j] = xn1;
        float r1 = fast_tanh(xn1);
        rbuf[nxt][0][lr * LDSP + col] = f2bf(r1);
        hid1[gidx] = r1;

        float xn2 = oma * x2[nti][j] + al * (c2a[nti][j] + c2b[nti][j]);
        x2[nti][j] = xn2;
        float r2 = fast_tanh(xn2);
        rbuf[nxt][1][lr * LDSP + col] = f2bf(r2);
        hid2[gidx] = r2;

        float xnb = oma * x1b[nti][j] + al * c1bb[nti][j];
        x1b[nti][j] = xnb;
        float rb = fast_tanh(xnb);
        rbuf[nxt][2][lr * LDSP + col] = f2bf(rb);
        hid1b[gidx] = rb;
      }
    }
    __syncthreads();   // one barrier/step: protects both r buffers (double-buffered)
  }
}

// ---------------------------------------------------------------------------
__global__ void cvt_kernel(const float* __restrict__ s, unsigned short* __restrict__ d,
                           int n) {
  int i = blockIdx.x * blockDim.x + threadIdx.x;
  if (i < n) d[i] = f2bf(s[i]);
}

// outv = hiddenl2 @ W_out.T + b_out   (bandwidth bound: 256 MB read)
__global__ void out_kernel(const float* __restrict__ hid2, const float* __restrict__ Wout,
                           const float* __restrict__ bout, float* __restrict__ outv) {
  int idx = blockIdx.x * blockDim.x + threadIdx.x;
  if (idx >= T_STEPS * BATCH) return;
  const float4* h4 = (const float4*)(hid2 + (size_t)idx * NH);
  const float4* w0 = (const float4*)(Wout);
  const float4* w1 = (const float4*)(Wout + NH);
  float a0 = 0.f, a1 = 0.f;
  #pragma unroll 4
  for (int k = 0; k < NH / 4; ++k) {
    float4 h = h4[k], p = w0[k], q = w1[k];
    a0 += h.x * p.x + h.y * p.y + h.z * p.z + h.w * p.w;
    a1 += h.x * q.x + h.y * q.y + h.z * q.z + h.w * q.w;
  }
  outv[idx * 2 + 0] = a0 + bout[0];
  outv[idx * 2 + 1] = a1 + bout[1];
}

// ---------------------------------------------------------------------------
extern "C" void kernel_launch(void* const* d_in, const int* in_sizes, int n_in,
                              void* d_out, int out_size, void* d_ws, size_t ws_size,
                              hipStream_t stream) {
  const float* X     = (const float*)d_in[0];
  const float* h01   = (const float*)d_in[1];
  const float* h02   = (const float*)d_in[2];
  const float* h01b  = (const float*)d_in[3];
  const float* Wih0  = (const float*)d_in[4];
  const float* Whh0  = (const float*)d_in[5];
  const float* Wih1  = (const float*)d_in[6];
  const float* Whh1  = (const float*)d_in[7];
  const float* Wih0b = (const float*)d_in[8];
  const float* Whh0b = (const float*)d_in[9];
  const float* W0b0  = (const float*)d_in[10];
  const float* Wout  = (const float*)d_in[11];
  const float* bout  = (const float*)d_in[12];

  unsigned short* ws = (unsigned short*)d_ws;   // 5 x 65536 bf16 = 640 KB
  const int NW = NH * NH;
  cvt_kernel<<<256, 256, 0, stream>>>(Whh0,  ws + 0 * NW, NW);
  cvt_kernel<<<256, 256, 0, stream>>>(W0b0,  ws + 1 * NW, NW);
  cvt_kernel<<<256, 256, 0, stream>>>(Whh0b, ws + 2 * NW, NW);
  cvt_kernel<<<256, 256, 0, stream>>>(Whh1,  ws + 3 * NW, NW);
  cvt_kernel<<<256, 256, 0, stream>>>(Wih1,  ws + 4 * NW, NW);

  float* out   = (float*)d_out;                        // [outv | h1 | h2 | h1b]
  float* outv  = out;
  float* hid1  = out  + (size_t)T_STEPS * BATCH * 2;
  float* hid2  = hid1 + (size_t)T_STEPS * BATCH * NH;
  float* hid1b = hid2 + (size_t)T_STEPS * BATCH * NH;

  ctrnn_kernel<<<16, 256, 0, stream>>>(X, h01, h02, h01b, Wih0, Wih0b, ws,
                                       hid1, hid2, hid1b);
  out_kernel<<<1000, 256, 0, stream>>>(hid2, Wout, bout, outv);
}
